// MultiHeadAttention_37666863185940
// MI455X (gfx1250) — matrix operand order repaired and checked by reading, not compiled, not run on hardware
//
#include <hip/hip_runtime.h>

typedef __attribute__((ext_vector_type(16))) _Float16 v16h;
typedef __attribute__((ext_vector_type(8)))  _Float16 v8h;
typedef __attribute__((ext_vector_type(4)))  _Float16 v4h;
typedef __attribute__((ext_vector_type(8)))  float    v8f;
typedef _Float16 f16;

#define EMBED 768
#define HEADS 12
#define HDIM 64
#define BATCH 4
#define SEQ 2048
#define MROWS (BATCH*SEQ)                  // 8192
#define HEAD_ELEMS (BATCH*HEADS*SEQ*HDIM)  // 6,291,456
#define W_ELEMS (EMBED*EMBED)              // 589,824
#define KITERS (EMBED/32)                  // 24

// ---------------------------------------------------------------------------
// WMMA fragment helpers (A frag in contiguous k = 16h + s order, same as B loads)
// ---------------------------------------------------------------------------
__device__ __forceinline__ v16h make_afrag(const f16* row, int kbase) {
    v8h lo = *(const v8h*)(row + 2 * kbase);
    v8h hi = *(const v8h*)(row + 2 * kbase + 8);
    return __builtin_shufflevector(lo, hi, 0,1,2,3,4,5,6,7,8,9,10,11,12,13,14,15);
}
__device__ __forceinline__ v8f wmma_f16(v16h a, v16h b, v8f c) {
    return __builtin_amdgcn_wmma_f32_16x16x32_f16(false, a, false, b, (short)0, c, false, false);
}

// ---------------------------------------------------------------------------
// CDNA5 async global->LDS copy (ASYNCcnt path, ISA §15.18.3 / §7 counters)
// ---------------------------------------------------------------------------
__device__ __forceinline__ unsigned lds_off(const void* p) {
    // generic pointer to __shared__: low 32 bits are the LDS offset (ISA 10.2)
    return (unsigned)(size_t)p;
}
__device__ __forceinline__ void async_b128(unsigned lds, const f16* g) {
    asm volatile("global_load_async_to_lds_b128 %0, %1, off"
                 :: "v"(lds), "v"((unsigned long long)(size_t)g)
                 : "memory");
}
__device__ __forceinline__ void wait_async_le3() {
    asm volatile("s_wait_asynccnt 0x3" ::: "memory");
}
__device__ __forceinline__ void wait_async_0() {
    asm volatile("s_wait_asynccnt 0x0" ::: "memory");
}

// ---------------------------------------------------------------------------
// Stage 0: fp32 -> f16 conversion (one-shot; halves GEMM read traffic and
// enables the async global->LDS pipeline)
// ---------------------------------------------------------------------------
__global__ void cvt_f32_to_f16(const float* __restrict__ s, f16* __restrict__ d,
                               int n4) {
    int i = blockIdx.x * 256 + threadIdx.x;
    const int stride = gridDim.x * 256;
    for (; i < n4; i += stride) {
        float4 f = ((const float4*)s)[i];
        v4h h; h[0] = (f16)f.x; h[1] = (f16)f.y; h[2] = (f16)f.z; h[3] = (f16)f.w;
        ((v4h*)d)[i] = h;
    }
}

// ---------------------------------------------------------------------------
// Shared double-buffered async GEMM mainloop:
//   acc[4](16x64 per wave) += A[64x768](f16) @ B[768x128]^T(f16) tile
// ---------------------------------------------------------------------------
__device__ __forceinline__ void gemm_mainloop(const f16* __restrict__ A,
                                              const f16* __restrict__ B,
                                              f16 (*As)[64 * 40],
                                              f16 (*Bs)[128 * 40],
                                              int mbase, int cbase,
                                              int t, int w, int lane,
                                              v8f acc[4]) {
    const int ln = lane & 15;
    const int kbA = (lane >> 4) * 8, koffB = (lane >> 4) * 16;
    const int ar = t & 63, ac = (t >> 6) * 8;   // A coop coords: 1x b128/thread
    const int br = t >> 1, bc = (t & 1) * 16;   // B coop coords: 2x b128/thread

    const f16* aSrc = A + (size_t)(mbase + ar) * EMBED + ac;
    const f16* bSrc = B + (size_t)(cbase + br) * EMBED + bc;
    const unsigned aL[2] = { lds_off(&As[0][ar * 40 + ac]),
                             lds_off(&As[1][ar * 40 + ac]) };
    const unsigned bL[2] = { lds_off(&Bs[0][br * 40 + bc]),
                             lds_off(&Bs[1][br * 40 + bc]) };

    // prologue: tile 0 -> buf 0  (3 async b128 per thread)
    async_b128(aL[0], aSrc);
    async_b128(bL[0], bSrc);
    async_b128(bL[0] + 16, bSrc + 8);

    for (int it = 0; it < KITERS; ++it) {
        const int buf = it & 1;
        if (it + 1 < KITERS) {
            const f16* a2 = aSrc + (it + 1) * 32;
            const f16* b2 = bSrc + (it + 1) * 32;
            async_b128(aL[buf ^ 1], a2);
            async_b128(bL[buf ^ 1], b2);
            async_b128(bL[buf ^ 1] + 16, b2 + 8);
            wait_async_le3();     // in-order: tile `it`'s 3 loads retired
        } else {
            wait_async_0();
        }
        __syncthreads();

        v16h af = make_afrag(&As[buf][((w >> 1) * 16 + ln) * 40], kbA);
        #pragma unroll
        for (int nt = 0; nt < 4; ++nt) {
            v16h bf = *(const v16h*)(&Bs[buf][((w & 1) * 64 + nt * 16 + ln) * 40 + koffB]);
            acc[nt] = wmma_f16(af, bf, acc[nt]);
        }
        __syncthreads();   // all waves done reading buf before it is re-filled
    }
}

// ---------------------------------------------------------------------------
// Stage 1: QKV projections (f16 inputs).  z selects Q/K/V.
// Q,K -> f16 [b,h,s,dh] ; V -> f16 transposed [b,h,dh,s].
// ---------------------------------------------------------------------------
__global__ __launch_bounds__(256, 2)
void mha_qkv_proj(const f16* __restrict__ xq, const f16* __restrict__ xk,
                  const f16* __restrict__ xv,
                  const f16* __restrict__ Wq, const f16* __restrict__ Wk,
                  const f16* __restrict__ Wv,
                  const float* __restrict__ bq, const float* __restrict__ bk,
                  const float* __restrict__ bv,
                  f16* __restrict__ qh, f16* __restrict__ kh, f16* __restrict__ vt)
{
    __shared__ f16 As[2][64 * 40];
    __shared__ f16 Bs[2][128 * 40];

    const int z = blockIdx.z;
    const f16* X = (z == 0) ? xq : (z == 1) ? xk : xv;
    const f16* W = (z == 0) ? Wq : (z == 1) ? Wk : Wv;
    const float* bias = (z == 0) ? bq : (z == 1) ? bk : bv;

    const int t = threadIdx.x, lane = t & 31, w = t >> 5;
    const int mbase = blockIdx.y * 64, cbase = blockIdx.x * 128;
    const int mwave = mbase + (w >> 1) * 16;
    const int cwave = cbase + (w & 1) * 64;
    const int ln = lane & 15, hi8 = (lane >> 4) * 8;

    v8f acc[4] = {v8f{}, v8f{}, v8f{}, v8f{}};
    gemm_mainloop(X, W, As, Bs, mbase, cbase, t, w, lane, acc);

    #pragma unroll
    for (int nt = 0; nt < 4; ++nt) {
        const int n = cwave + nt * 16 + ln;       // 0..767
        const float bv_ = bias[n];
        const int h = n >> 6, dh = n & 63;
        if (z < 2) {
            f16* dst = (z == 0) ? qh : kh;
            #pragma unroll
            for (int r = 0; r < 8; ++r) {
                const int m = mwave + r + hi8;
                const int b = m >> 11, s = m & 2047;
                dst[((size_t)(b * HEADS + h) * SEQ + s) * HDIM + dh] =
                    (f16)(acc[nt][r] + bv_);
            }
        } else {
            const int sbase = mwave + hi8;
            const int b = sbase >> 11, s = sbase & 2047;
            v8h pv;
            #pragma unroll
            for (int r = 0; r < 8; ++r) pv[r] = (f16)(acc[nt][r] + bv_);
            *(v8h*)(&vt[((size_t)(b * HEADS + h) * HDIM + dh) * SEQ + s]) = pv;
        }
    }
}

// ---------------------------------------------------------------------------
// Stage 2: attention.  One WG per (bh, 16-query block).
// Scores (16x2048 fp32) entirely in the 320KB WGP LDS -> exact softmax,
// attn written once to HBM, P kept in LDS as f16, P @ V via WMMA split-K.
// ---------------------------------------------------------------------------
#define SC_LD 2052
#define P_LD  2064

__global__ __launch_bounds__(256, 1)
void mha_attention(const f16* __restrict__ qh, const f16* __restrict__ kh,
                   const f16* __restrict__ vt,
                   float* __restrict__ attn, f16* __restrict__ ctx)
{
    __shared__ float sc[16 * SC_LD];
    __shared__ f16   p16[16 * P_LD];
    __shared__ float red[4 * 256];

    const int t = threadIdx.x, lane = t & 31, w = t >> 5;
    const int ln = lane & 15, hi8 = (lane >> 4) * 8;
    const int kbA = (lane >> 4) * 8, koffB = (lane >> 4) * 16;
    const int qbase = blockIdx.x * 16, bh = blockIdx.y;
    const int b = bh / HEADS, h = bh % HEADS;
    const float scale = 0.125f;   // 1/sqrt(64)

    const f16* qrow = qh + ((size_t)bh * SEQ + qbase + ln) * HDIM;
    v16h qa0 = make_afrag(qrow, kbA);
    v16h qa1 = make_afrag(qrow + 32, kbA);

    // Phase 1: scores
    for (int kt = w; kt < 128; kt += 8) {
        const f16* krow = kh + ((size_t)bh * SEQ + kt * 16 + ln) * HDIM;
        v16h kb0 = *(const v16h*)(krow + koffB);
        v16h kb1 = *(const v16h*)(krow + 32 + koffB);
        v8f s = {};
        s = wmma_f16(qa0, kb0, s);
        s = wmma_f16(qa1, kb1, s);
        #pragma unroll
        for (int r = 0; r < 8; ++r)
            sc[(r + hi8) * SC_LD + kt * 16 + ln] = s[r] * scale;
    }
    __syncthreads();

    // Phase 2: exact softmax; write attn (fp32, coalesced) + P (f16, LDS)
    #pragma unroll
    for (int rr = w * 2; rr < w * 2 + 2; ++rr) {
        float mx = -3.0e38f;
        for (int c = lane; c < SEQ; c += 32) mx = fmaxf(mx, sc[rr * SC_LD + c]);
        #pragma unroll
        for (int off = 16; off; off >>= 1) mx = fmaxf(mx, __shfl_xor(mx, off, 32));
        float sum = 0.f;
        for (int c = lane; c < SEQ; c += 32) {
            float e = __expf(sc[rr * SC_LD + c] - mx);
            sc[rr * SC_LD + c] = e;
            sum += e;
        }
        #pragma unroll
        for (int off = 16; off; off >>= 1) sum += __shfl_xor(sum, off, 32);
        const float inv = 1.0f / sum;
        float* arow = attn + ((size_t)bh * SEQ + qbase + rr) * SEQ;
        for (int c = lane; c < SEQ; c += 32) {
            float a = sc[rr * SC_LD + c] * inv;
            arow[c] = a;
            p16[rr * P_LD + c] = (f16)a;
        }
    }
    __syncthreads();

    // Phase 3: O = P @ V (split-K across wave pairs, LDS reduce)
    const int nt = w & 3, khalf = w >> 2;
    v8f o = {};
    const f16* vrow = vt + ((size_t)bh * HDIM + nt * 16 + ln) * SEQ;
    for (int kc = khalf * 1024; kc < khalf * 1024 + 1024; kc += 32) {
        v16h pa = make_afrag(&p16[ln * P_LD + kc], kbA);
        v16h vb = *(const v16h*)(vrow + kc + koffB);
        o = wmma_f16(pa, vb, o);
    }
    if (khalf == 1) {
        #pragma unroll
        for (int r = 0; r < 8; ++r) red[nt * 256 + lane * 8 + r] = o[r];
    }
    __syncthreads();
    if (khalf == 0) {
        #pragma unroll
        for (int r = 0; r < 8; ++r) {
            float v = o[r] + red[nt * 256 + lane * 8 + r];
            const int m = r + hi8;
            const size_t tok = (size_t)b * SEQ + qbase + m;
            ctx[tok * EMBED + h * HDIM + nt * 16 + ln] = (f16)v;
        }
    }
}

// ---------------------------------------------------------------------------
// Stage 3: output projection. out[8192x768] = ctx(f16) @ Wo^T(f16) + bo (fp32)
// ---------------------------------------------------------------------------
__global__ __launch_bounds__(256, 2)
void mha_out_proj(const f16* __restrict__ ctx, const f16* __restrict__ Wo,
                  const float* __restrict__ bo, float* __restrict__ out)
{
    __shared__ f16 As[2][64 * 40];
    __shared__ f16 Bs[2][128 * 40];

    const int t = threadIdx.x, lane = t & 31, w = t >> 5;
    const int mbase = blockIdx.y * 64, cbase = blockIdx.x * 128;
    const int mwave = mbase + (w >> 1) * 16;
    const int cwave = cbase + (w & 1) * 64;
    const int ln = lane & 15, hi8 = (lane >> 4) * 8;

    v8f acc[4] = {v8f{}, v8f{}, v8f{}, v8f{}};
    gemm_mainloop(ctx, Wo, As, Bs, mbase, cbase, t, w, lane, acc);

    #pragma unroll
    for (int nt = 0; nt < 4; ++nt) {
        const int n = cwave + nt * 16 + ln;
        const float bias = bo[n];
        #pragma unroll
        for (int r = 0; r < 8; ++r) {
            const int m = mwave + r + hi8;
            out[(size_t)m * EMBED + n] = acc[nt][r] + bias;
        }
    }
}

// ---------------------------------------------------------------------------
extern "C" void kernel_launch(void* const* d_in, const int* in_sizes, int n_in,
                              void* d_out, int out_size, void* d_ws, size_t ws_size,
                              hipStream_t stream) {
    (void)in_sizes; (void)n_in; (void)out_size; (void)ws_size;
    const float* query = (const float*)d_in[0];
    const float* key   = (const float*)d_in[1];
    const float* value = (const float*)d_in[2];
    const float* Wq    = (const float*)d_in[3];
    const float* bq    = (const float*)d_in[4];
    const float* Wk    = (const float*)d_in[5];
    const float* bk    = (const float*)d_in[6];
    const float* Wv    = (const float*)d_in[7];
    const float* bv    = (const float*)d_in[8];
    const float* Wo    = (const float*)d_in[9];
    const float* bo    = (const float*)d_in[10];

    float* out  = (float*)d_out;
    float* attn = out + (size_t)BATCH * SEQ * EMBED;

    f16* ws = (f16*)d_ws;
    size_t o = 0;
    f16* qh   = ws + o; o += (size_t)HEAD_ELEMS;   // [b,h,s,dh]
    f16* kh   = ws + o; o += (size_t)HEAD_ELEMS;   // [b,h,s,dh]
    f16* vt   = ws + o; o += (size_t)HEAD_ELEMS;   // [b,h,dh,s]
    f16* ctx  = ws + o; o += (size_t)HEAD_ELEMS;   // [b,s,e]
    f16* xq16 = ws + o; o += (size_t)HEAD_ELEMS;
    f16* xk16 = ws + o; o += (size_t)HEAD_ELEMS;
    f16* xv16 = ws + o; o += (size_t)HEAD_ELEMS;
    f16* wq16 = ws + o; o += (size_t)W_ELEMS;
    f16* wk16 = ws + o; o += (size_t)W_ELEMS;
    f16* wv16 = ws + o; o += (size_t)W_ELEMS;
    f16* wo16 = ws + o; o += (size_t)W_ELEMS;

    // Stage 0: one-shot fp32 -> f16 conversions
    const int xq4 = HEAD_ELEMS / 4, wq4 = W_ELEMS / 4;
    cvt_f32_to_f16<<<dim3(xq4 / 256), 256, 0, stream>>>(query, xq16, xq4);
    cvt_f32_to_f16<<<dim3(xq4 / 256), 256, 0, stream>>>(key,   xk16, xq4);
    cvt_f32_to_f16<<<dim3(xq4 / 256), 256, 0, stream>>>(value, xv16, xq4);
    cvt_f32_to_f16<<<dim3(wq4 / 256), 256, 0, stream>>>(Wq, wq16, wq4);
    cvt_f32_to_f16<<<dim3(wq4 / 256), 256, 0, stream>>>(Wk, wk16, wq4);
    cvt_f32_to_f16<<<dim3(wq4 / 256), 256, 0, stream>>>(Wv, wv16, wq4);
    cvt_f32_to_f16<<<dim3(wq4 / 256), 256, 0, stream>>>(Wo, wo16, wq4);

    mha_qkv_proj<<<dim3(EMBED / 128, MROWS / 64, 3), 256, 0, stream>>>(
        xq16, xk16, xv16, wq16, wk16, wv16, bq, bk, bv, qh, kh, vt);

    mha_attention<<<dim3(SEQ / 16, BATCH * HEADS), 256, 0, stream>>>(
        qh, kh, vt, attn, ctx);

    mha_out_proj<<<dim3(EMBED / 128, MROWS / 64), 256, 0, stream>>>(
        ctx, wo16, bo, out);
}
